// GConvLSTM_38173669327257
// MI455X (gfx1250) — matrix operand order repaired and checked by reading, not compiled, not run on hardware
//
#include <hip/hip_runtime.h>

// ---------------------------------------------------------------------------
// GConvLSTM single step for MI455X (gfx1250).
// Refactor: S@(x@W) == (S@x)@W  -> aggregate 128-ch features once, then do
// all 8 gate matmuls with bf16 WMMA, fused with the LSTM pointwise math.
// ---------------------------------------------------------------------------

#define N_NODES 50000
#define N_EDGES 1600000
#define CH      128   // IN_C == OUT_C

typedef __attribute__((ext_vector_type(16))) __bf16 bf16x16;
typedef __attribute__((ext_vector_type(8)))  float  f32x8;

__device__ __forceinline__ unsigned short f2bf_rne(float f) {
  unsigned int b = __float_as_uint(f);
  b += 0x7FFFu + ((b >> 16) & 1u);          // round-to-nearest-even
  return (unsigned short)(b >> 16);
}

__device__ __forceinline__ bf16x16 pack16(float4 a, float4 b, float4 c, float4 d) {
  union { bf16x16 v; unsigned short u[16]; } t;
  t.u[0]  = f2bf_rne(a.x); t.u[1]  = f2bf_rne(a.y);
  t.u[2]  = f2bf_rne(a.z); t.u[3]  = f2bf_rne(a.w);
  t.u[4]  = f2bf_rne(b.x); t.u[5]  = f2bf_rne(b.y);
  t.u[6]  = f2bf_rne(b.z); t.u[7]  = f2bf_rne(b.w);
  t.u[8]  = f2bf_rne(c.x); t.u[9]  = f2bf_rne(c.y);
  t.u[10] = f2bf_rne(c.z); t.u[11] = f2bf_rne(c.w);
  t.u[12] = f2bf_rne(d.x); t.u[13] = f2bf_rne(d.y);
  t.u[14] = f2bf_rne(d.z); t.u[15] = f2bf_rne(d.w);
  return t.v;
}

__device__ __forceinline__ float fast_sigmoid(float x) {
  return 1.0f / (1.0f + __expf(-x));
}
__device__ __forceinline__ float fast_tanh(float x) {
  float t = __expf(-2.0f * x);
  return (1.0f - t) / (1.0f + t);
}

// ---- 1. zero scratch (deg + dis + AX + AH region) -------------------------
__global__ void k_zero(float* p, long long n) {
  long long i = (long long)blockIdx.x * blockDim.x + threadIdx.x;
  if (i < n) p[i] = 0.0f;
}

// ---- 2. in-degree (weighted) ---------------------------------------------
__global__ void k_degree(const int* __restrict__ ei,
                         const float* __restrict__ ew,
                         float* __restrict__ deg) {
  int e = blockIdx.x * blockDim.x + threadIdx.x;
  if (e >= N_EDGES) return;
  int dst = ei[N_EDGES + e];
  atomicAdd(&deg[dst], ew[e]);
}

// ---- 3. dis = deg>0 ? rsqrt(deg) : 0 --------------------------------------
__global__ void k_dis(const float* __restrict__ deg, float* __restrict__ dis) {
  int i = blockIdx.x * blockDim.x + threadIdx.x;
  if (i >= N_NODES) return;
  float d = deg[i];
  dis[i] = (d > 0.0f) ? __frsqrt_rn(d) : 0.0f;
}

// ---- 4a. pack 8 weight matrices into WMMA B-fragment layout (bf16) --------
// Fragment id (s,g,kt,j): s=0 Wx / s=1 Wh, gate g, k-tile kt (32), col-tile j (16).
// Element (lane, e): B[k][c] with k = kt*32 + (lane>>4)*8 + e (e<8) or kt*32 + 16 + (lane>>4)*8 + (e-8) (e>=8), c = j*16 + (lane&15).
__global__ void k_pack_w(const float* __restrict__ Wxi, const float* __restrict__ Wxf,
                         const float* __restrict__ Wxc, const float* __restrict__ Wxo,
                         const float* __restrict__ Whi, const float* __restrict__ Whf,
                         const float* __restrict__ Whc, const float* __restrict__ Who,
                         __bf16* __restrict__ Wb) {
  int tid = blockIdx.x * blockDim.x + threadIdx.x;   // 131072 total
  if (tid >= 2 * 4 * 4 * 8 * 32 * 16) return;
  int e    =  tid        & 15;
  int lane = (tid >> 4)  & 31;
  int j    = (tid >> 9)  & 7;
  int kt   = (tid >> 12) & 3;
  int g    = (tid >> 14) & 3;
  int s    = (tid >> 16) & 1;
  int k = kt * 32 + (lane >> 4) * 8 + (e & 7) + (e >> 3) * 16;
  int c = j * 16 + (lane & 15);
  const float* W;
  if (s == 0) W = (g == 0) ? Wxi : (g == 1) ? Wxf : (g == 2) ? Wxc : Wxo;
  else        W = (g == 0) ? Whi : (g == 1) ? Whf : (g == 2) ? Whc : Who;
  union { __bf16 h; unsigned short u; } cv;
  cv.u = f2bf_rne(W[k * CH + c]);
  Wb[tid] = cv.h;
}

// ---- 4b. combined bias: bsum[g*128+c] = bx_g + bh_g + b_g -----------------
__global__ void k_bias(const float* bxi, const float* bxf, const float* bxc, const float* bxo,
                       const float* bhi, const float* bhf, const float* bhc, const float* bho,
                       const float* bi,  const float* bf_, const float* bc,  const float* bo,
                       float* __restrict__ bsum) {
  int t = blockIdx.x * blockDim.x + threadIdx.x;
  if (t >= 512) return;
  int g = t >> 7, c = t & 127;
  const float *bx, *bh, *bb;
  if (g == 0)      { bx = bxi; bh = bhi; bb = bi; }
  else if (g == 1) { bx = bxf; bh = bhf; bb = bf_; }
  else if (g == 2) { bx = bxc; bh = bhc; bb = bc; }
  else             { bx = bxo; bh = bho; bb = bo; }
  bsum[t] = bx[c] + bh[c] + bb[c];
}

// ---- 5. edge gather/scale/scatter: AX += norm*X[src], AH += norm*H[src] ---
// One wave (32 lanes) per edge; each lane handles 4 channels (float4).
__global__ __launch_bounds__(256) void k_edge_agg(
    const int* __restrict__ ei, const float* __restrict__ ew,
    const float* __restrict__ dis,
    const float* __restrict__ X, const float* __restrict__ H,
    float* __restrict__ AX, float* __restrict__ AH) {
  int e = blockIdx.x * 8 + (threadIdx.x >> 5);
  if (e >= N_EDGES) return;
  int lane = threadIdx.x & 31;
  int src = ei[e];
  int dst = ei[N_EDGES + e];
  float nrm = dis[src] * ew[e] * dis[dst];
  const float4 xv = *(const float4*)(X + (size_t)src * CH + lane * 4);
  const float4 hv = *(const float4*)(H + (size_t)src * CH + lane * 4);
  float* ax = AX + (size_t)dst * CH + lane * 4;
  float* ah = AH + (size_t)dst * CH + lane * 4;
  atomicAdd(ax + 0, xv.x * nrm); atomicAdd(ax + 1, xv.y * nrm);
  atomicAdd(ax + 2, xv.z * nrm); atomicAdd(ax + 3, xv.w * nrm);
  atomicAdd(ah + 0, hv.x * nrm); atomicAdd(ah + 1, hv.y * nrm);
  atomicAdd(ah + 2, hv.z * nrm); atomicAdd(ah + 3, hv.w * nrm);
}

// ---- 6. fused WMMA gate matmul + LSTM pointwise ---------------------------
// Block = 256 threads = 8 waves. Block handles a 16-node tile; wave w handles
// channel tile j=w (16 channels) for ALL FOUR gates:
//   acc_g = sum_kt AXb(16x32) @ Wx_g_frag + AHb(16x32) @ Wh_g_frag
// Then: I,F sigmoid w/ peephole on C, T tanh, Cn, O sigmoid w/ peephole on Cn,
// Hn = O*tanh(Cn). Writes Hn and Cn directly to d_out.
__global__ __launch_bounds__(256) void k_gates(
    const float* __restrict__ AX, const float* __restrict__ AH,
    const __bf16* __restrict__ Wb, const float* __restrict__ bsum,
    const float* __restrict__ Cin,
    const float* __restrict__ wci, const float* __restrict__ wcf,
    const float* __restrict__ wco,
    float* __restrict__ Hn, float* __restrict__ Cn) {
  const int j    = threadIdx.x >> 5;       // column tile 0..7
  const int lane = threadIdx.x & 31;
  const int h    = lane >> 4;              // half-wave select
  const int n    = lane & 15;
  const int mbase = blockIdx.x * 16;

  f32x8 acc[4] = {f32x8{}, f32x8{}, f32x8{}, f32x8{}};

#pragma unroll
  for (int s = 0; s < 2; ++s) {
    const float* A = s ? AH : AX;
#pragma unroll
    for (int kt = 0; kt < 4; ++kt) {
      // A fragment (16x32 bf16): lane holds row M = n, half h;
      // k-chunks [kt*32 + 8h, +8) and [kt*32 + 16 + 8h, +8), each contiguous.
      const float* arow = A + (size_t)(mbase + n) * CH + kt * 32 + h * 8;
      float4 a0 = *(const float4*)(arow);
      float4 a1 = *(const float4*)(arow + 4);
      float4 a2 = *(const float4*)(arow + 16);
      float4 a3 = *(const float4*)(arow + 20);
      bf16x16 av = pack16(a0, a1, a2, a3);
#pragma unroll
      for (int g = 0; g < 4; ++g) {
        const bf16x16 bv = *(const bf16x16*)(
            Wb + ((((size_t)s * 4 + g) * 4 + kt) * 8 + j) * 512 + lane * 16);
        acc[g] = __builtin_amdgcn_wmma_f32_16x16x32_bf16(
            false, av, false, bv, (short)0, acc[g], false, false);
      }
    }
  }

  // ---- pointwise LSTM on the 16x16 tile held in acc[] ---------------------
  const int chan = j * 16 + n;
  const float bI = bsum[0 * CH + chan];
  const float bF = bsum[1 * CH + chan];
  const float bT = bsum[2 * CH + chan];
  const float bO = bsum[3 * CH + chan];
  const float wI = wci[chan], wF = wcf[chan], wO = wco[chan];

#pragma unroll
  for (int r = 0; r < 8; ++r) {
    const int node = mbase + r + 8 * h;    // C/D layout: M = r + 8*(lane/16)
    const size_t idx = (size_t)node * CH + chan;
    const float Cv = Cin[idx];
    const float I = fast_sigmoid(acc[0][r] + bI + wI * Cv);
    const float F = fast_sigmoid(acc[1][r] + bF + wF * Cv);
    const float T = fast_tanh(acc[2][r] + bT);
    const float Cnew = F * Cv + I * T;
    const float O = fast_sigmoid(acc[3][r] + bO + wO * Cnew);
    Hn[idx] = O * fast_tanh(Cnew);
    Cn[idx] = Cnew;
  }
}

// ---------------------------------------------------------------------------
extern "C" void kernel_launch(void* const* d_in, const int* in_sizes, int n_in,
                              void* d_out, int out_size, void* d_ws, size_t ws_size,
                              hipStream_t stream) {
  // inputs (setup_inputs dict order)
  const float* X   = (const float*)d_in[0];
  const int*   ei  = (const int*)  d_in[1];
  const float* ew  = (const float*)d_in[2];
  const float* H   = (const float*)d_in[3];
  const float* C   = (const float*)d_in[4];
  const float* Wx[4] = {(const float*)d_in[5],  (const float*)d_in[9],
                        (const float*)d_in[13], (const float*)d_in[17]};
  const float* bx[4] = {(const float*)d_in[6],  (const float*)d_in[10],
                        (const float*)d_in[14], (const float*)d_in[18]};
  const float* Wh[4] = {(const float*)d_in[7],  (const float*)d_in[11],
                        (const float*)d_in[15], (const float*)d_in[19]};
  const float* bh[4] = {(const float*)d_in[8],  (const float*)d_in[12],
                        (const float*)d_in[16], (const float*)d_in[20]};
  const float* wci = (const float*)d_in[21];
  const float* wcf = (const float*)d_in[22];
  const float* wco = (const float*)d_in[23];
  const float* bg[4] = {(const float*)d_in[24], (const float*)d_in[25],
                        (const float*)d_in[26], (const float*)d_in[27]};

  // workspace layout (floats)
  float* ws   = (float*)d_ws;
  float* deg  = ws;                                   // N
  float* dis  = deg + N_NODES;                        // N
  float* AX   = dis + N_NODES;                        // N*128
  float* AH   = AX + (size_t)N_NODES * CH;            // N*128
  float* bsum = AH + (size_t)N_NODES * CH;            // 512
  __bf16* Wb  = (__bf16*)(bsum + 512);                // 131072 bf16

  float* Hn = (float*)d_out;
  float* Cn = Hn + (size_t)N_NODES * CH;

  // 1. zero deg (+dis) + AX + AH
  long long zn = (long long)2 * N_NODES + 2LL * N_NODES * CH;
  k_zero<<<(int)((zn + 255) / 256), 256, 0, stream>>>(ws, zn);

  // 2. weighted in-degree
  k_degree<<<(N_EDGES + 255) / 256, 256, 0, stream>>>(ei, ew, deg);

  // 3. inverse sqrt degree
  k_dis<<<(N_NODES + 255) / 256, 256, 0, stream>>>(deg, dis);

  // 4. weight packing + bias folding (tiny)
  k_pack_w<<<131072 / 256, 256, 0, stream>>>(Wx[0], Wx[1], Wx[2], Wx[3],
                                             Wh[0], Wh[1], Wh[2], Wh[3], Wb);
  k_bias<<<2, 256, 0, stream>>>(bx[0], bx[1], bx[2], bx[3],
                                bh[0], bh[1], bh[2], bh[3],
                                bg[0], bg[1], bg[2], bg[3], bsum);

  // 5. sparse aggregation (bandwidth/atomic bound; dominant cost)
  k_edge_agg<<<(N_EDGES + 7) / 8, 256, 0, stream>>>(ei, ew, dis, X, H, AX, AH);

  // 6. fused WMMA gate matmuls + LSTM pointwise
  k_gates<<<N_NODES / 16, 256, 0, stream>>>(AX, AH, Wb, bsum, C,
                                            wci, wcf, wco, Hn, Cn);
}
